// Encoder_69449621176639
// MI455X (gfx1250) — compile-verified, matrix operand order checked by reading, not run on hardware
//
#include <hip/hip_runtime.h>
#include <cstdint>

// LSTM: B=64, T=512, I=256, H=512, gates 4H=2048, fused K = I+H = 768.
#define BDIM 64
#define TDIM 512
#define IDIM 256
#define HDIM 512
#define GDIM 2048
#define KDIM 768
#define NKB  24     // KDIM / 32
#define NNB  128    // GDIM / 16
#define NWG  16     // persistent workgroups

typedef __attribute__((ext_vector_type(16))) __bf16 v16bf;
typedef __attribute__((ext_vector_type(8)))  float  v8f;

struct b32x8 { uint4 lo, hi; };

static __device__ __forceinline__ unsigned short f32_to_bf16(float f) {
    union { float f; unsigned u; } v; v.f = f;
    unsigned u = v.u;
    unsigned r = u + 0x7FFFu + ((u >> 16) & 1u);
    return (unsigned short)(r >> 16);
}
static __device__ __forceinline__ float sigmoid_fast(float x) {
    return 1.0f / (1.0f + __expf(-x));
}
static __device__ __forceinline__ float tanh_fast(float x) {
    return 2.0f / (1.0f + __expf(-2.0f * x)) - 1.0f;
}

// Per-lane K pattern of bf16 WMMA A/B fragments (ISA §7.12.2):
//   kpat(lane, r, e) = (r<4 ? 0 : 16) + (lane>=16 ? 8 : 0) + 2*(r&3) + e

// ---------------------------------------------------------------------------
// Pack [W_ih | W_hh]^T (f32) -> bf16 B-fragment layout:
//   [nb(128)][kb(24)][lane(32)][r(8)][e(2)], 512 bf16 (1 KB) per fragment.
// Gate columns interleaved: packed col p = 4*j + gate.
// ---------------------------------------------------------------------------
__global__ void lstm_pack(const float* __restrict__ Wih,
                          const float* __restrict__ Whh,
                          unsigned short* __restrict__ wtp) {
    int gid = blockIdx.x * blockDim.x + threadIdx.x;
    if (gid >= KDIM * GDIM) return;
    int e    =  gid        & 1;
    int r    = (gid >> 1)  & 7;
    int lane = (gid >> 4)  & 31;
    int fi   =  gid >> 9;
    int kb   = fi % NKB;
    int nb   = fi / NKB;
    int p    = nb * 16 + (lane & 15);
    int j    = p >> 2;
    int gate = p & 3;
    int row  = gate * HDIM + j;
    int kpat = ((r < 4) ? 0 : 16) + ((lane >= 16) ? 8 : 0) + 2 * (r & 3) + e;
    int k    = kb * 32 + kpat;
    float v  = (k < IDIM) ? Wih[(size_t)row * IDIM + k]
                          : Whh[(size_t)row * HDIM + (k - IDIM)];
    wtp[gid] = f32_to_bf16(v);
}

// ---------------------------------------------------------------------------
// Pack x (f32 [B,T,I]) -> bf16 A-fragment layout per timestep:
//   [t(512)][mt(4)][kbx(8)][lane(32)][r(8)][e(2)]   (16 MB, bandwidth-bound)
// ---------------------------------------------------------------------------
__global__ void lstm_xpack(const float* __restrict__ x,
                           unsigned short* __restrict__ xp) {
    int gid = blockIdx.x * blockDim.x + threadIdx.x;  // 2^23 elements
    if (gid >= TDIM * 4 * 8 * 512) return;
    int e    =  gid        & 1;
    int r    = (gid >> 1)  & 7;
    int lane = (gid >> 4)  & 31;
    int kbx  = (gid >> 9)  & 7;
    int mt   = (gid >> 12) & 3;
    int t    =  gid >> 14;
    int b    = mt * 16 + (lane & 15);
    int kpat = ((r < 4) ? 0 : 16) + ((lane >= 16) ? 8 : 0) + 2 * (r & 3) + e;
    int k    = kbx * 32 + kpat;
    xp[gid] = f32_to_bf16(x[((size_t)b * TDIM + t) * IDIM + k]);
}

// Zero h-pack ping buffer, build permuted bias, reset the grid barrier.
__global__ void lstm_init(unsigned short* __restrict__ hp0,
                          float* __restrict__ biasPerm,
                          unsigned* __restrict__ bar,
                          const float* __restrict__ bih,
                          const float* __restrict__ bhh) {
    int gid = blockIdx.x * blockDim.x + threadIdx.x;
    if (gid < BDIM * HDIM) hp0[gid] = 0;          // 32768 bf16 zeros
    if (gid < GDIM) {
        int j = gid >> 2, gate = gid & 3;
        biasPerm[gid] = bih[gate * HDIM + j] + bhh[gate * HDIM + j];
    }
    if (gid == 0) *bar = 0u;
}

// ---------------------------------------------------------------------------
// Persistent LSTM: 16 WGs x 8 waves. Wave (wg,w) owns gate columns
// [ (wg*8+w)*16 , +16 ), keeps all 24 K-fragments of W^T in VGPRs (192 VGPRs),
// and computes 4 row-tiles (full batch of 64). A-fragments are explicitly
// double-buffered so loads for kb+1 overlap the WMMAs of kb. Cell state c
// stays in registers for the entire sequence. Steps are separated by a grid
// barrier implemented with a monotonic device-scope atomic counter.
// ---------------------------------------------------------------------------
__global__ void __launch_bounds__(256, 1)
lstm_persistent(const unsigned short* __restrict__ xp,
                const unsigned short* __restrict__ wtp,
                const float* __restrict__ biasPerm,
                unsigned short* __restrict__ hp0,
                unsigned short* __restrict__ hp1,
                float* __restrict__ hidden,
                float* __restrict__ hn,
                float* __restrict__ cnout,
                unsigned* __restrict__ bar) {
    __shared__ float Gs[64 * 128];   // 32 KB fp32 gates tile
    __shared__ float Bs[128];        // this WG's 128 gate-column biases

    const int tid  = threadIdx.x;
    const int wg   = blockIdx.x;      // 0..15
    const int w    = tid >> 5;        // wave 0..7
    const int lane = tid & 31;
    const int sel  = lane >> 4;
    const int ml   = lane & 15;
    const int nb   = wg * 8 + w;      // global 16-column block 0..127

    if (tid < 128) Bs[tid] = biasPerm[wg * 128 + tid];

    // --- Preload all B fragments for this wave's columns into VGPRs ---
    v16bf bfr[NKB];
    #pragma unroll
    for (int kb = 0; kb < NKB; ++kb) {
        const unsigned short* bp = wtp + (((size_t)nb * NKB + kb) << 9) + lane * 16;
        b32x8 raw;
        raw.lo = *(const uint4*)(bp);
        raw.hi = *(const uint4*)(bp + 8);
        bfr[kb] = __builtin_bit_cast(v16bf, raw);
    }

    float creg[8];
    #pragma unroll
    for (int i = 0; i < 8; ++i) creg[i] = 0.0f;

    for (int t = 0; t < TDIM; ++t) {
        const unsigned short* hin  = (t & 1) ? hp1 : hp0;
        unsigned short*       hout = (t & 1) ? hp0 : hp1;

        v8f acc[4];
        v16bf afA[4], afB[4];

        const unsigned short* xbase = xp + (((size_t)t * 32) << 9) + lane * 16;

        // prologue: load kb=0 A fragments
        #pragma unroll
        for (int mt = 0; mt < 4; ++mt) {
            const unsigned short* ap = xbase + (((size_t)mt * 8) << 9);
            b32x8 raw;
            raw.lo = *(const uint4*)(ap);
            raw.hi = *(const uint4*)(ap + 8);
            afA[mt] = __builtin_bit_cast(v16bf, raw);
        }

        #pragma unroll
        for (int kb = 0; kb < NKB; ++kb) {
            v16bf* cur = (kb & 1) ? afB : afA;
            v16bf* nxt = (kb & 1) ? afA : afB;
            if (kb + 1 < NKB) {
                const int kn = kb + 1;
                #pragma unroll
                for (int mt = 0; mt < 4; ++mt) {
                    const unsigned short* ap = (kn < 8)
                        ? xbase + (((size_t)mt * 8 + kn) << 9)
                        : hin + (((size_t)mt * 16 + (kn - 8)) << 9) + lane * 16;
                    b32x8 raw;
                    raw.lo = *(const uint4*)(ap);
                    raw.hi = *(const uint4*)(ap + 8);
                    nxt[mt] = __builtin_bit_cast(v16bf, raw);
                }
            }
            #pragma unroll
            for (int mt = 0; mt < 4; ++mt) {
                if (kb == 0) {
                    // first K-block: inline-zero C operand (no init copies)
                    acc[mt] = __builtin_amdgcn_wmma_f32_16x16x32_bf16(
                        false, cur[mt], false, bfr[kb], (short)0, (v8f){},
                        false, false);
                } else {
                    acc[mt] = __builtin_amdgcn_wmma_f32_16x16x32_bf16(
                        false, cur[mt], false, bfr[kb], (short)0, acc[mt],
                        false, false);
                }
            }
        }
        if (t + 1 < TDIM)   // warm caches for next step's x fragments
            __builtin_prefetch((const void*)(xp + (((size_t)(t + 1) * 32) << 9) + lane * 16), 0, 0);

        // --- scatter gates into LDS (C layout: VGPR v -> row v + 8*sel) ---
        #pragma unroll
        for (int mt = 0; mt < 4; ++mt) {
            #pragma unroll
            for (int v = 0; v < 8; ++v)
                Gs[(mt * 16 + v + 8 * sel) * 128 + w * 16 + ml] = acc[mt][v];
        }
        __syncthreads();

        // --- LSTM cell update: 64 batch x 32 hidden units per WG ---
        #pragma unroll
        for (int i = 0; i < 8; ++i) {
            int e  = tid + i * 256;          // time-invariant thread mapping
            int b  = e >> 5;
            int jl = e & 31;
            int c0 = jl * 4;
            float4 g4 = *(const float4*)&Gs[b * 128 + c0];   // ds_load_b128
            float4 b4 = *(const float4*)&Bs[c0];
            float ig = sigmoid_fast(g4.x + b4.x);
            float fg = sigmoid_fast(g4.y + b4.y);
            float gt = tanh_fast   (g4.z + b4.z);
            float og = sigmoid_fast(g4.w + b4.w);
            float cnew = fg * creg[i] + ig * gt;
            float h    = og * tanh_fast(cnew);
            creg[i] = cnew;

            int j = wg * 32 + jl;
            hidden[((size_t)b * TDIM + t) * HDIM + j] = h;

            // scatter h (bf16) into packed A-fragment layout for next step
            int klocal = j & 31;
            int kbh    = j >> 5;
            int selh   = (klocal >> 3) & 1;
            int rr     = ((klocal & 16) >> 2) + ((klocal & 7) >> 1);
            int ee     = klocal & 1;
            int laneh  = (b & 15) + (selh << 4);
            int mth    = b >> 4;
            hout[(((size_t)mth * 16 + kbh) << 9) + laneh * 16 + rr * 2 + ee] =
                f32_to_bf16(h);

            if (t == TDIM - 1) {
                int ci = b * HDIM + j;
                hn[ci]    = h;
                cnout[ci] = cnew;
            }
        }
        __syncthreads();

        // --- grid barrier: monotonic counter, one arrival per WG ---
        if (tid == 0) {
            __threadfence();                 // release h/hidden writes
            atomicAdd(bar, 1u);
            unsigned target = (unsigned)NWG * (unsigned)(t + 1);
            while (__hip_atomic_load(bar, __ATOMIC_ACQUIRE,
                                     __HIP_MEMORY_SCOPE_AGENT) < target)
                __builtin_amdgcn_s_sleep(1);
        }
        __syncthreads();
    }
}

// ---------------------------------------------------------------------------
extern "C" void kernel_launch(void* const* d_in, const int* in_sizes, int n_in,
                              void* d_out, int out_size, void* d_ws, size_t ws_size,
                              hipStream_t stream) {
    const float* x   = (const float*)d_in[0];
    const float* Wih = (const float*)d_in[1];
    const float* Whh = (const float*)d_in[2];
    const float* bih = (const float*)d_in[3];
    const float* bhh = (const float*)d_in[4];

    float* hidden = (float*)d_out;                         // [B,T,H]
    float* hn     = hidden + (size_t)BDIM * TDIM * HDIM;   // [1,B,H]
    float* cn     = hn + (size_t)BDIM * HDIM;              // [1,B,H]

    // Workspace (~19.3 MB)
    char* ws = (char*)d_ws;
    unsigned short* wtp = (unsigned short*)ws;  ws += (size_t)KDIM * GDIM * 2;        // 3 MB
    unsigned short* xp  = (unsigned short*)ws;  ws += (size_t)TDIM * BDIM * IDIM * 2; // 16 MB
    float* biasPerm = (float*)ws;               ws += (size_t)GDIM * 4;
    unsigned short* hp0 = (unsigned short*)ws;  ws += (size_t)BDIM * HDIM * 2;
    unsigned short* hp1 = (unsigned short*)ws;  ws += (size_t)BDIM * HDIM * 2;
    unsigned* bar = (unsigned*)ws;              ws += 256;

    lstm_pack <<<(KDIM * GDIM + 255) / 256, 256, 0, stream>>>(Wih, Whh, wtp);
    lstm_xpack<<<(TDIM * BDIM * IDIM + 255) / 256, 256, 0, stream>>>(x, xp);
    lstm_init <<<(BDIM * HDIM + 255) / 256, 256, 0, stream>>>(hp0, biasPerm, bar, bih, bhh);

    lstm_persistent<<<NWG, 256, 0, stream>>>(xp, wtp, biasPerm, hp0, hp1,
                                             hidden, hn, cn, bar);
}
